// MultiHeadAttention_26259430047904
// MI455X (gfx1250) — compile-verified, matrix operand order checked by reading, not run on hardware
//
#include <hip/hip_runtime.h>
#include <hip/hip_bf16.h>

// ---------------------------------------------------------------------------
// MI455X (gfx1250) fused multi-head attention, bf16 WMMA with fp32 accum.
//   B=2, S=2048, H=2048, NH=16, HD=128  (hardcoded, matches reference)
// Pipeline:
//   1) cast X, Wq, Wk, Wv, Wo  fp32 -> bf16            (cast_bf16_k)
//   2) Q/K = X @ W^T + b  -> bf16 [B,NH,S,HD]          (gemm_xwt_k<1>)
//      V   = X @ Wv^T + b -> bf16 [B,NH,HD,S] d-major  (gemm_xwt_k<2>)
//   3) flash attention (causal) -> ctx bf16 [B,S,H]    (flash_attn_k)
//   4) out = ctx @ Wo^T + bo   -> fp32 [B,S,H]         (gemm_xwt_k<0>)
// CDNA5 paths: v_wmma_f32_16x16x32_bf16 compute; TDM tensor_load_to_lds
// (6-arg amdgpu-toolchain form, TENSORcnt) for GEMM tile staging;
// global_load_async_to_lds_b128 (ASYNCcnt) for attention tile staging.
// ---------------------------------------------------------------------------

typedef __bf16 bf16;
typedef __attribute__((ext_vector_type(16))) bf16  v16bf;
typedef __attribute__((ext_vector_type(8)))  float v8f;
typedef int v4i __attribute__((ext_vector_type(4)));

constexpr int Bc  = 2;
constexpr int Sc  = 2048;
constexpr int Hc  = 2048;
constexpr int NHc = 16;
constexpr int HDc = 128;
constexpr int Mc  = Bc * Sc;     // 4096 token rows

// ---------------------------------------------------------------------------
// CDNA5 feature probes
// ---------------------------------------------------------------------------
#if __has_builtin(__builtin_amdgcn_tensor_load_to_lds) && \
    __has_builtin(__builtin_amdgcn_s_wait_tensorcnt)
#define HAVE_TDM 1
#else
#define HAVE_TDM 0
#endif

#if __has_builtin(__builtin_amdgcn_global_load_async_to_lds_b128)
#define HAVE_ASYNC 1
#else
#define HAVE_ASYNC 0
#endif

#if HAVE_ASYNC
typedef __attribute__((address_space(1))) v4i* gptr_v4i;   // global-side 128b
typedef __attribute__((address_space(3))) v4i* lptr_v4i;   // LDS-side 128b
#endif

#if HAVE_TDM
// Raw 32-bit LDS byte offset of a generic pointer into shared memory.
__device__ __forceinline__ unsigned lds_byte_off(void* p) {
  return (unsigned)(unsigned long long)(__attribute__((address_space(3))) char*)p;
}

typedef unsigned int u32x4v __attribute__((ext_vector_type(4)));
typedef int          i32x8v __attribute__((ext_vector_type(8)));
typedef int          i32x4v __attribute__((ext_vector_type(4)));

// DMA a 128-row x 32-col bf16 tile (row stride 2048 elems) from global into
// LDS via the Tensor Data Mover. 2-D descriptor per CDNA5 ISA:
//  group0: count=1 | lds_addr | global_addr[56:0] | type=2
//  group1: data_size=1(2B) | tensor_dim0=2048 | tensor_dim1=rows |
//          tile_dim0=32 | tile_dim1=128 | tensor_dim0_stride=2048
__device__ __forceinline__ void tdm_load_tile_128x32(const bf16* gsrc,
                                                     unsigned lds_addr,
                                                     unsigned tensor_rows) {
  unsigned long long ga = (unsigned long long)gsrc;
  u32x4v g0;
  g0[0] = 1u;                                                 // count=1
  g0[1] = lds_addr;                                           // lds_addr
  g0[2] = (unsigned)(ga & 0xffffffffu);                       // global_addr lo
  g0[3] = (unsigned)((ga >> 32) & 0x01ffffffu) | (2u << 30);  // addr hi | type=2
  i32x8v g1;
  g1[0] = 0x00010000;                          // workgroup_mask=0, data_size=1 (2B)
  g1[1] = (int)(2048u << 16);                  // tensor_dim0[15:0]=2048 (bits 63:48)
  g1[2] = (int)(tensor_rows << 16);            // tensor_dim1[15:0]     (bits 95:80)
  g1[3] = (int)(32u << 16);                    // tile_dim0=32          (bits 127:112)
  g1[4] = (int)128;                            // tile_dim1=128         (bits 143:128)
  g1[5] = (int)2048;                           // tensor_dim0_stride=2048
  g1[6] = 0;
  g1[7] = 0;
  i32x4v z4 = {0, 0, 0, 0};
  i32x8v z8 = {0, 0, 0, 0, 0, 0, 0, 0};
  __builtin_amdgcn_tensor_load_to_lds(g0, g1, z4, z4, z8, 0);  // 6-arg form
}
#endif

// Load a 16x32 bf16 WMMA fragment (A-style: 16 rows across lanes, 32 K along
// vector elems; lanes 0-15 hold K 0..7 & 16..23, lanes 16-31 hold 8..15 & 24..31).
// Used for both A and B operands (B assumed symmetric: lane = N column).
__device__ __forceinline__ v16bf ld_frag(const bf16* t, int stride) {
  const int l = threadIdx.x & 31;
  const bf16* p = t + (l & 15) * stride + ((l >> 4) << 3);
  union { unsigned int u[8]; v16bf v; } fr;
  *(uint4*)(&fr.u[0]) = *(const uint4*)p;         // K+0..7   (16B)
  *(uint4*)(&fr.u[4]) = *(const uint4*)(p + 16);  // K+16..23 (16B)
  return fr.v;
}

// ---------------------------------------------------------------------------
// fp32 -> bf16 cast, float4-vectorized
// ---------------------------------------------------------------------------
__global__ void cast_bf16_k(const float* __restrict__ s, bf16* __restrict__ d, int n4) {
  int i = blockIdx.x * blockDim.x + threadIdx.x;
  if (i < n4) {
    float4 f = ((const float4*)s)[i];
    __align__(8) bf16 h[4];
    h[0] = (bf16)f.x; h[1] = (bf16)f.y; h[2] = (bf16)f.z; h[3] = (bf16)f.w;
    *(uint2*)(d + 4 * (size_t)i) = *(const uint2*)h;
  }
}

// ---------------------------------------------------------------------------
// C = A[M,K] @ W[N,K]^T + bias   (bf16 in, fp32 accum)
// Block tile 128x128, 8 waves of 32, wave tile 64x32 (4x2 WMMA 16x16 tiles).
// LAYOUT 0: fp32 [M,N].  LAYOUT 1: bf16 [B,NH,S,HD].  LAYOUT 2: bf16 [B,NH,HD,S].
// ---------------------------------------------------------------------------
template <int LAYOUT>
__global__ __launch_bounds__(256) void gemm_xwt_k(const bf16* __restrict__ A,
                                                  const bf16* __restrict__ W,
                                                  const float* __restrict__ bias,
                                                  float* __restrict__ outF,
                                                  bf16* __restrict__ outH) {
  constexpr int Kc = Hc;   // 2048
  constexpr int Nc = Hc;   // 2048
  __shared__ __align__(16) bf16 smA[128 * 32];
  __shared__ __align__(16) bf16 smB[128 * 32];

  const int m0 = blockIdx.y * 128, n0 = blockIdx.x * 128;
  const int w = threadIdx.x >> 5, lane = threadIdx.x & 31;
  const int ln = lane & 15, hi = lane >> 4;
  const int wm = (w >> 2) * 64;   // 0 / 64
  const int wn = (w & 3) * 32;    // 0 / 32 / 64 / 96

  v8f acc[4][2] = {};

  for (int k0 = 0; k0 < Kc; k0 += 32) {
    __syncthreads();
#if HAVE_TDM
    // Tensor Data Mover stages both 128x32 tiles; wave 0 issues (EXEC-ignored
    // whole-wave op), waits TENSORcnt, then workgroup barrier releases all.
    if (w == 0) {
      if (lane == 0) {
        tdm_load_tile_128x32(&A[(size_t)m0 * Kc + k0], lds_byte_off(smA), (unsigned)Mc);
        tdm_load_tile_128x32(&W[(size_t)n0 * Kc + k0], lds_byte_off(smB), (unsigned)Nc);
      }
      __builtin_amdgcn_s_wait_tensorcnt(0);
    }
#else
    // Stage 128x32 bf16 tiles of A and W into LDS (2 x uint4 per thread each).
#pragma unroll
    for (int it = 0; it < 2; ++it) {
      int idx = threadIdx.x + it * 256;       // 512 uint4 per tile
      int r = idx >> 2, c = (idx & 3) << 3;
      *(uint4*)&smA[r * 32 + c] = *(const uint4*)&A[(size_t)(m0 + r) * Kc + k0 + c];
      *(uint4*)&smB[r * 32 + c] = *(const uint4*)&W[(size_t)(n0 + r) * Kc + k0 + c];
    }
    if (k0 + 32 < Kc) {   // prefetch next K tile (global_prefetch_b8)
      __builtin_prefetch(&A[(size_t)(m0 + (threadIdx.x >> 1)) * Kc + k0 + 32 + (threadIdx.x & 1) * 16], 0, 1);
      __builtin_prefetch(&W[(size_t)(n0 + (threadIdx.x >> 1)) * Kc + k0 + 32 + (threadIdx.x & 1) * 16], 0, 1);
    }
#endif
    __syncthreads();

    v16bf afr[4], bfr[2];
#pragma unroll
    for (int mi = 0; mi < 4; ++mi) afr[mi] = ld_frag(&smA[(wm + mi * 16) * 32], 32);
#pragma unroll
    for (int ni = 0; ni < 2; ++ni) bfr[ni] = ld_frag(&smB[(wn + ni * 16) * 32], 32);
#pragma unroll
    for (int mi = 0; mi < 4; ++mi)
#pragma unroll
      for (int ni = 0; ni < 2; ++ni)
        acc[mi][ni] = __builtin_amdgcn_wmma_f32_16x16x32_bf16(
            false, afr[mi], false, bfr[ni], (short)0, acc[mi][ni], false, false);
  }

  // Epilogue: C/D layout -> VGPR i holds row (i | i+8), column = lane&15.
#pragma unroll
  for (int mi = 0; mi < 4; ++mi)
#pragma unroll
    for (int ni = 0; ni < 2; ++ni)
#pragma unroll
      for (int i = 0; i < 8; ++i) {
        int row = m0 + wm + mi * 16 + i + hi * 8;
        int col = n0 + wn + ni * 16 + ln;
        float v = acc[mi][ni][i] + bias[col];
        if (LAYOUT == 0) {
          outF[(size_t)row * Nc + col] = v;
        } else {
          int bb = row >> 11, s = row & (Sc - 1);    // row over B*S
          int hh = col >> 7,  d = col & (HDc - 1);   // col over H
          if (LAYOUT == 1)
            outH[(((size_t)(bb * NHc + hh)) * Sc + s) * HDc + d] = (bf16)v;
          else  // LAYOUT 2: d-major V for transpose-free attention staging
            outH[(((size_t)(bb * NHc + hh)) * HDc + d) * Sc + s] = (bf16)v;
        }
      }
}

// ---------------------------------------------------------------------------
// Causal flash attention. Grid: (S/128, NH, B), 256 threads (8 waves).
// Each wave owns 16 query rows; Q fragments in registers; K tile + V^T tile
// staged in dynamic LDS (async-to-LDS when available); per-wave LDS staging
// for P (C-frag -> A-frag relayout).
// ---------------------------------------------------------------------------
__global__ __launch_bounds__(256) void flash_attn_k(const bf16* __restrict__ Qb,
                                                    const bf16* __restrict__ Kb,
                                                    const bf16* __restrict__ Vt,
                                                    bf16* __restrict__ Ct) {
  extern __shared__ char smem[];
  bf16* smK  = (bf16*)smem;          // [128][136]  keys x head-dim
  bf16* smVt = smK + 128 * 136;      // [128][136]  head-dim x keys (d-major)
  bf16* smP  = smVt + 128 * 136;     // [8 waves][16][128] probs staging

  const int qt = blockIdx.x;         // query tile index
  const int h  = blockIdx.y;
  const int b  = blockIdx.z;
  const int q0 = qt * 128;
  const int w = threadIdx.x >> 5, lane = threadIdx.x & 31;
  const int ln = lane & 15, hi = lane >> 4;

  const size_t headBase = ((size_t)(b * NHc + h)) * Sc * HDc;  // same for K and Vt

  // Q fragments: wave rows q0+w*16 .. +15, four K-chunks of 32 over HD=128.
  v16bf qa[4];
  {
    const bf16* qrow = Qb + headBase + (size_t)(q0 + w * 16 + ln) * HDc;
#pragma unroll
    for (int kk = 0; kk < 4; ++kk) {
      union { unsigned int u[8]; v16bf v; } fr;
      const bf16* p = qrow + kk * 32 + hi * 8;
      *(uint4*)(&fr.u[0]) = *(const uint4*)p;
      *(uint4*)(&fr.u[4]) = *(const uint4*)(p + 16);
      qa[kk] = fr.v;
    }
  }

  v8f o[8] = {};                       // 16 rows x 128 head-dim, fp32
  float m_run[8], l_run[8];
#pragma unroll
  for (int i = 0; i < 8; ++i) { m_run[i] = -3.0e38f; l_run[i] = 0.0f; }

  const float scale = 0.08838834764831845f;  // 1/sqrt(128)

  for (int kt = 0; kt <= qt; ++kt) {         // causal: only lower-tri key tiles
    __syncthreads();                         // protect smK/smVt reuse
    // Cooperative staging: both tiles are straight 16B copies now that V is
    // stored d-major by its projection GEMM.
#pragma unroll
    for (int it = 0; it < 8; ++it) {
      int idx = threadIdx.x + it * 256;      // 2048 uint4 per tile
      int r  = idx >> 4;                     // smK: key row / smVt: d row
      int cb = (idx & 15) << 3;              // 8-elem column block
      const bf16* gk = Kb + headBase + (size_t)(kt * 128 + r) * HDc + cb;
      const bf16* gv = Vt + headBase + (size_t)r * Sc + kt * 128 + cb;
#if HAVE_ASYNC
      __builtin_amdgcn_global_load_async_to_lds_b128(
          (gptr_v4i)(void*)gk, (lptr_v4i)(void*)&smK[r * 136 + cb], 0, 0);
      __builtin_amdgcn_global_load_async_to_lds_b128(
          (gptr_v4i)(void*)gv, (lptr_v4i)(void*)&smVt[r * 136 + cb], 0, 0);
#else
      *(uint4*)&smK[r * 136 + cb]  = *(const uint4*)gk;
      *(uint4*)&smVt[r * 136 + cb] = *(const uint4*)gv;
#endif
    }
#if HAVE_ASYNC
#if __has_builtin(__builtin_amdgcn_s_wait_asynccnt)
    __builtin_amdgcn_s_wait_asynccnt(0);
#else
    asm volatile("s_wait_asynccnt 0x0" ::: "memory");
#endif
#endif
    __syncthreads();

    // S = Q @ K^T  (8 key sub-tiles x 4 HD chunks of 32)
    v8f sfr[8];
#pragma unroll
    for (int nt = 0; nt < 8; ++nt) {
      v8f sa = {};
#pragma unroll
      for (int kk = 0; kk < 4; ++kk) {
        v16bf bf = ld_frag(&smK[(nt * 16) * 136 + kk * 32], 136);
        sa = __builtin_amdgcn_wmma_f32_16x16x32_bf16(
            false, qa[kk], false, bf, (short)0, sa, false, false);
      }
      sfr[nt] = sa;
    }

    // Scale + causal mask (in-register).
#pragma unroll
    for (int nt = 0; nt < 8; ++nt)
#pragma unroll
      for (int i = 0; i < 8; ++i) {
        int col = kt * 128 + nt * 16 + ln;
        int row = q0 + w * 16 + i + hi * 8;
        float v = sfr[nt][i] * scale;
        sfr[nt][i] = (col > row) ? -1.0e30f : v;
      }

    // Online softmax per row slot i (rows i / i+8 live in lane halves).
#pragma unroll
    for (int i = 0; i < 8; ++i) {
      float mx = m_run[i];
#pragma unroll
      for (int nt = 0; nt < 8; ++nt) mx = fmaxf(mx, sfr[nt][i]);
#pragma unroll
      for (int off = 1; off < 16; off <<= 1) mx = fmaxf(mx, __shfl_xor(mx, off, 32));
      float al = expf(m_run[i] - mx);
      float rs = 0.0f;
#pragma unroll
      for (int nt = 0; nt < 8; ++nt) {
        float p = expf(sfr[nt][i] - mx);
        rs += p;
        smP[w * 2048 + (i + hi * 8) * 128 + nt * 16 + ln] = (bf16)p;
      }
#pragma unroll
      for (int off = 1; off < 16; off <<= 1) rs += __shfl_xor(rs, off, 32);
      l_run[i] = l_run[i] * al + rs;
      m_run[i] = mx;
#pragma unroll
      for (int nt = 0; nt < 8; ++nt) o[nt][i] *= al;
    }

    // O += P @ V  (per-wave smP region; DS ops in-order within wave)
#pragma unroll
    for (int kk = 0; kk < 4; ++kk) {
      v16bf pa = ld_frag(&smP[w * 2048 + kk * 32], 128);
#pragma unroll
      for (int nt = 0; nt < 8; ++nt) {
        v16bf bv = ld_frag(&smVt[(nt * 16) * 136 + kk * 32], 136);
        o[nt] = __builtin_amdgcn_wmma_f32_16x16x32_bf16(
            false, pa, false, bv, (short)0, o[nt], false, false);
      }
    }
  }

  // Normalize and store ctx (token-major [B,S,H], bf16).
#pragma unroll
  for (int i = 0; i < 8; ++i) {
    float inv = 1.0f / l_run[i];
    int srow = q0 + w * 16 + i + hi * 8;
#pragma unroll
    for (int nt = 0; nt < 8; ++nt) {
      int d = nt * 16 + ln;
      Ct[((size_t)(b * Sc + srow)) * Hc + h * HDc + d] = (bf16)(o[nt][i] * inv);
    }
  }
}

// ---------------------------------------------------------------------------
// Host launcher
// ---------------------------------------------------------------------------
extern "C" void kernel_launch(void* const* d_in, const int* in_sizes, int n_in,
                              void* d_out, int out_size, void* d_ws, size_t ws_size,
                              hipStream_t stream) {
  const float* hs = (const float*)d_in[0];
  // d_in[1] = causal_mask (unused; causality handled analytically)
  const float* Wq = (const float*)d_in[2]; const float* bq = (const float*)d_in[3];
  const float* Wk = (const float*)d_in[4]; const float* bk = (const float*)d_in[5];
  const float* Wv = (const float*)d_in[6]; const float* bv = (const float*)d_in[7];
  const float* Wo = (const float*)d_in[8]; const float* bo = (const float*)d_in[9];
  float* out = (float*)d_out;

  const size_t nX = (size_t)Mc * Hc;   // 8M elems
  const size_t nW = (size_t)Hc * Hc;   // 4M elems

  bf16* Xh  = (bf16*)d_ws;       // bf16 workspace partitions (~112 MB total)
  bf16* Wqh = Xh  + nX;
  bf16* Wkh = Wqh + nW;
  bf16* Wvh = Wkh + nW;
  bf16* Woh = Wvh + nW;
  bf16* Qb  = Woh + nW;
  bf16* Kb  = Qb  + nX;
  bf16* Vtb = Kb  + nX;
  bf16* Ct  = Vtb + nX;

  cast_bf16_k<<<(int)(nX / 4 / 256), 256, 0, stream>>>(hs, Xh, (int)(nX / 4));
  cast_bf16_k<<<(int)(nW / 4 / 256), 256, 0, stream>>>(Wq, Wqh, (int)(nW / 4));
  cast_bf16_k<<<(int)(nW / 4 / 256), 256, 0, stream>>>(Wk, Wkh, (int)(nW / 4));
  cast_bf16_k<<<(int)(nW / 4 / 256), 256, 0, stream>>>(Wv, Wvh, (int)(nW / 4));
  cast_bf16_k<<<(int)(nW / 4 / 256), 256, 0, stream>>>(Wo, Woh, (int)(nW / 4));

  dim3 gg(Hc / 128, Mc / 128);   // (16, 32)
  gemm_xwt_k<1><<<gg, 256, 0, stream>>>(Xh, Wqh, bq, nullptr, Qb);
  gemm_xwt_k<1><<<gg, 256, 0, stream>>>(Xh, Wkh, bk, nullptr, Kb);
  gemm_xwt_k<2><<<gg, 256, 0, stream>>>(Xh, Wvh, bv, nullptr, Vtb);  // d-major V

  size_t smem = (size_t)(2 * 128 * 136 + 8 * 16 * 128) * sizeof(bf16);  // ~100 KB
  flash_attn_k<<<dim3(Sc / 128, NHc, Bc), 256, smem, stream>>>(Qb, Kb, Vtb, Ct);

  gemm_xwt_k<0><<<gg, 256, 0, stream>>>(Ct, Woh, bo, out, nullptr);
}
